// Model_21320217658146
// MI455X (gfx1250) — matrix operand order repaired and checked by reading, not compiled, not run on hardware
//
#include <hip/hip_runtime.h>

// ---------------------------------------------------------------------------
// MS-G3D-ish skeleton net forward on gfx1250 (CDNA5).
// All 1x1 convs and 3-tap temporal convs run as bf16 WMMA GEMMs
// (v_wmma_f32_16x16x32_bf16), fp32 accumulate, fused bias/BN/res/ReLU epilogue.
// B operands stored spatial-major (S x Kp) bf16 -> staging is one aligned
// b128 load per thread with immediate offsets. Operands pre-padded so the
// k-loop is branch-free. Block tile 64x64, 8 wave32, 2 WMMAs per wave per
// k-step. LDS double-buffered + register prefetch: one barrier per k-step,
// global latency hidden behind the WMMAs.
// ---------------------------------------------------------------------------

typedef unsigned short u16;
typedef unsigned int   u32;
typedef __attribute__((ext_vector_type(16))) __bf16 v16bf;
typedef __attribute__((ext_vector_type(8)))  float  v8f;

#define GF_RELU     1
#define GF_ACC      2   // add old output AFTER bn
#define GF_BN       4
#define GF_BIAS     8
#define GF_ACCFIRST 16  // add old output BEFORE bn (multi-tap accumulation)

#define LSTRIDE 40      // LDS row stride (u16): 16B-aligned b128 stores,
                        // 20-uint stride -> bank-conflict-free frag reads

__device__ __forceinline__ u16 f2bf(float f) {
    union { float f; u32 u; } x; x.f = f;
    u32 u = x.u;
    return (u16)((u + 0x7FFFu + ((u >> 16) & 1u)) >> 16);
}

// ---------------------------------------------------------------------------
// WMMA GEMM:  Out[m, s] (+)= act( W[m,:] . X[col(s), :] )
//   W : bf16 (Mp64 x Kp) row-major, zero-padded
//   X : bf16 (S_inP x Kp) spatial-major per sample; pad cols zero, margin rows
//       zero.  col(s): s=(t,v) -> (t*t_stride + t_off_m)*25 + v
// ---------------------------------------------------------------------------
__global__ __launch_bounds__(256) void gemm_wmma_kernel(
    const u16* __restrict__ W, const u16* __restrict__ X, float* __restrict__ Out,
    int Mdim, int Kp,
    int T_out, int t_stride, int t_off_m,
    long x_ss, long out_ss,
    const float* __restrict__ bias, const float* __restrict__ bns,
    const float* __restrict__ bnb, int flags)
{
    const int Vd    = 25;
    const int S_out = T_out * Vd;

    __shared__ u16 lA[2][64 * LSTRIDE];
    __shared__ u16 lB[2][64 * LSTRIDE];
    __shared__ int cmap[64];

    const int tid = threadIdx.x;
    const int n0  = blockIdx.x * 64;
    const int m0  = blockIdx.y * 64;
    const int z   = blockIdx.z;
    const u16* Xs = X + (long)z * x_ss;
    float*     Os = Out + (long)z * out_ss;

    if (tid < 64) {
        int s = n0 + tid;
        if (s >= S_out) s = 0;               // clamp: load row 0, discard later
        int t = s / Vd, v = s - t * Vd;
        cmap[tid] = (t * t_stride + t_off_m) * Vd + v;
    }
    __syncthreads();

    const int lane = tid & 31;
    const int wv   = tid >> 5;   // 0..7
    const int wm   = wv >> 1;    // 0..3 (16-row strip)
    const int wn   = wv & 1;     // 0..1 (32-col strip)
    const int half = lane >> 4;
    const int l16  = lane & 15;

    const int arow = tid >> 2;          // 0..63
    const int akc  = (tid & 3) * 8;     // 0,8,16,24
    const int bcol = tid & 63;          // 0..63
    const int bk0  = (tid >> 6) * 8;    // 0,8,16,24

    const u16* wrow = W + (long)(m0 + arow) * Kp + akc;        // contiguous in k
    const u16* xrow = Xs + (long)cmap[bcol] * Kp + bk0;        // contiguous in k
    const int aoff = arow * LSTRIDE + akc;
    const int boff = bcol * LSTRIDE + bk0;

    v8f acc0 = {}, acc1 = {};

    // prefetch first tiles into registers
    uint4 aq = *(const uint4*)(wrow);
    uint4 bq = *(const uint4*)(xrow);

    int buf = 0;
    for (int k0 = 0; k0 < Kp; k0 += 32) {
        // ---- commit prefetched tile to LDS buffer `buf` ----
        *(uint4*)&lA[buf][aoff] = aq;
        *(uint4*)&lB[buf][boff] = bq;
        __syncthreads();

        // ---- prefetch next tile (latency hidden behind WMMAs) ----
        if (k0 + 32 < Kp) {
            aq = *(const uint4*)(wrow + k0 + 32);
            bq = *(const uint4*)(xrow + k0 + 32);
        }

        // ---- fragments per documented CDNA5 bf16 layouts ----
        union { v16bf v; u32 u[8]; } af, bf0, bf1;
        {
            const u16* ap = &lA[buf][(wm * 16 + l16) * LSTRIDE];
            const int kb = half * 8;
            #pragma unroll
            for (int j = 0; j < 4; ++j) af.u[j] = *(const u32*)(ap + kb + 2 * j);
            #pragma unroll
            for (int j = 4; j < 8; ++j) af.u[j] = *(const u32*)(ap + 16 + kb + 2 * (j - 4));
            const u16* bp0 = &lB[buf][(wn * 32 + l16) * LSTRIDE + half * 8];
            const u16* bp1 = bp0 + 16 * LSTRIDE;
            #pragma unroll
            for (int j = 0; j < 8; ++j) bf0.u[j] = *(const u32*)(bp0 + 2 * j + (j >= 4 ? 8 : 0));
            #pragma unroll
            for (int j = 0; j < 8; ++j) bf1.u[j] = *(const u32*)(bp1 + 2 * j + (j >= 4 ? 8 : 0));
        }
        acc0 = __builtin_amdgcn_wmma_f32_16x16x32_bf16(
                   false, af.v, false, bf0.v, (short)0, acc0, false, false);
        acc1 = __builtin_amdgcn_wmma_f32_16x16x32_bf16(
                   false, af.v, false, bf1.v, (short)0, acc1, false, false);
        buf ^= 1;   // next writes go to the other buffer: no trailing barrier
    }

    // ---- fused epilogue (two 16x16 tiles per wave) ----
    #pragma unroll
    for (int tile = 0; tile < 2; ++tile) {
        union { v8f v; float f[8]; } cf;
        cf.v = tile ? acc1 : acc0;
        int col = n0 + wn * 32 + tile * 16 + l16;
        if (col < S_out) {
            #pragma unroll
            for (int r = 0; r < 8; ++r) {
                int m = m0 + wm * 16 + half * 8 + r;
                if (m < Mdim) {
                    float v = cf.f[r];
                    if (flags & GF_BIAS) v += bias[m];
                    long oidx = (long)m * S_out + col;
                    float old = 0.f;
                    if (flags & (GF_ACC | GF_ACCFIRST)) old = Os[oidx];
                    if (flags & GF_ACCFIRST) v += old;
                    if (flags & GF_BN) v = bns[m] * v + bnb[m];
                    if ((flags & GF_ACC) && !(flags & GF_ACCFIRST)) v += old;
                    if (flags & GF_RELU) v = v > 0.f ? v : 0.f;
                    Os[oidx] = v;
                }
            }
        }
    }
}

// ---------------------------------------------------------------------------
// data_bn + (N,C,T,V,M) -> (N*M, C, T, V) permute, fp32 output
// ---------------------------------------------------------------------------
__global__ void databn_kernel(const float* __restrict__ x, const float* __restrict__ s,
                              const float* __restrict__ b, float* __restrict__ of)
{
    int i = blockIdx.x * 256 + threadIdx.x;
    if (i >= 16 * 3 * 300 * 25) return;
    int v = i % 25;
    int t = (i / 25) % 300;
    int c = (i / (25 * 300)) % 3;
    int nm = i / (3 * 300 * 25);
    int n = nm >> 1, m = nm & 1;
    float val = x[((((long)n * 3 + c) * 300 + t) * 25 + v) * 2 + m];
    int ch = (m * 25 + v) * 3 + c;
    of[i] = val * s[ch] + b[ch];
}

// graph aggregation (one sample), transposed output (S x Kp), pad cols zero:
// aggT[t*25+w][k*C+c] = sum_u (Apow+Ares)[kV+w,u] * x[c,t,u]
__global__ void aggT_kernel(const float* __restrict__ Apow, const float* __restrict__ Ares,
                            const float* __restrict__ xs, u16* __restrict__ agg,
                            int C, int T, int Kp)
{
    int total = Kp * T * 25;
    int i = blockIdx.x * 256 + threadIdx.x;
    if (i >= total) return;
    int kc = i % Kp, s = i / Kp;
    if (kc >= 13 * C) { agg[i] = 0; return; }
    int c = kc % C, k = kc / C;
    int t = s / 25, w = s - t * 25;
    const float* ar = Apow + (k * 25 + w) * 25;
    const float* rr = Ares + (k * 25 + w) * 25;
    const float* xr = xs + ((long)c * T + t) * 25;
    float acc = 0.f;
    #pragma unroll
    for (int u = 0; u < 25; ++u) acc += (ar[u] + rr[u]) * xr[u];
    agg[i] = f2bf(acc);
}

// weights fp32 -> padded bf16 (Mp64 x Kp); kstride/koff pick a temporal tap
__global__ void packW_kernel(const float* __restrict__ W, u16* __restrict__ P,
                             int Md, int Kd, int Mp, int Kp, int kstride, int koff)
{
    int i = blockIdx.x * 256 + threadIdx.x;
    if (i >= Mp * Kp) return;
    int m = i / Kp, k = i - m * Kp;
    float v = (m < Md && k < Kd) ? W[((long)m * Kd + k) * kstride + koff] : 0.f;
    P[i] = f2bf(v);
}

// activations fp32 (16, C, S) -> bf16 transposed (16, S, C); C multiple of 32
__global__ void packAT_kernel(const float* __restrict__ X, u16* __restrict__ Y,
                              int C, int S)
{
    long total = 16L * S * C;
    long i = (long)blockIdx.x * 256 + threadIdx.x;
    if (i >= total) return;
    int c = (int)(i % C); long r = i / C;
    int s = (int)(r % S); int z = (int)(r / S);
    Y[i] = f2bf(X[((long)z * C + c) * S + s]);
}

// branch tmp fp32 (16, bc, T*25) -> bf16 transposed (16, (T+8)*25, KpT) with
// zeroed K-pad cols and +-4 timestep margin rows (branch-free dilated taps)
__global__ void pack_padT_kernel(const float* __restrict__ src, u16* __restrict__ dst,
                                 int bc, int KpT, int T)
{
    int SP = (T + 8) * 25;
    long total = 16L * KpT * SP;
    long i = (long)blockIdx.x * 256 + threadIdx.x;
    if (i >= total) return;
    int row = (int)(i % KpT); long r = i / KpT;
    int p = (int)(r % SP); int z = (int)(r / SP);
    u16 val = 0;
    int q = p - 100;            // margin = 4 timesteps * 25
    if (row < bc && q >= 0 && q < T * 25)
        val = f2bf(src[((long)z * bc + row) * (T * 25) + q]);
    dst[i] = val;
}

// maxpool3 over T (pad -inf) + BN; H:(16,bc,T_in,25) -> Out slice:(16,bc,T_out,25)
__global__ void maxpool_bn_kernel(const float* __restrict__ H, float* __restrict__ Out,
                                  const float* __restrict__ s, const float* __restrict__ b,
                                  int bc, int T_in, int T_out, int stride,
                                  long hss, long oss)
{
    long total = 16L * bc * T_out * 25;
    long i = (long)blockIdx.x * 256 + threadIdx.x;
    if (i >= total) return;
    int v = (int)(i % 25); long r = i / 25;
    int t = (int)(r % T_out); r /= T_out;
    int c = (int)(r % bc); int z = (int)(r / bc);
    const float* hz = H + z * hss + (long)c * T_in * 25;
    int tc = t * stride;
    float m = -3.4e38f;
    #pragma unroll
    for (int dt = -1; dt <= 1; ++dt) {
        int ti = tc + dt;
        if (ti >= 0 && ti < T_in) { float x = hz[(long)ti * 25 + v]; m = x > m ? x : m; }
    }
    Out[z * oss + ((long)c * T_out + t) * 25 + v] = s[c] * m + b[c];
}

__global__ void add_relu_kernel(float* __restrict__ Out, const float* __restrict__ Res,
                                long n, int relu)
{
    long i = (long)blockIdx.x * 256 + threadIdx.x;
    if (i >= n) return;
    float v = Out[i];
    if (Res) v += Res[i];
    if (relu) v = v > 0.f ? v : 0.f;
    Out[i] = v;
}

// global mean over (M, T, V) per (n, c): (16,384,75,25) -> (8,384)
__global__ void meanpool_kernel(const float* __restrict__ H, float* __restrict__ pooled,
                                int C, int S)
{
    int c = blockIdx.x % C, n = blockIdx.x / C;
    float s = 0.f;
    for (int i = threadIdx.x; i < 2 * S; i += 256) {
        int m = i / S; int off = i - m * S;
        s += H[(((long)(n * 2 + m)) * C + c) * S + off];
    }
    __shared__ float red[256];
    red[threadIdx.x] = s; __syncthreads();
    for (int w = 128; w > 0; w >>= 1) {
        if (threadIdx.x < w) red[threadIdx.x] += red[threadIdx.x + w];
        __syncthreads();
    }
    if (threadIdx.x == 0) pooled[n * C + c] = red[0] / (float)(2 * S);
}

__global__ void fc_kernel(const float* __restrict__ pooled, const float* __restrict__ W,
                          const float* __restrict__ b, float* __restrict__ out)
{
    int i = blockIdx.x * 64 + threadIdx.x;
    if (i >= 480) return;
    int o = i % 60, n = i / 60;
    float s = b[o];
    const float* p = pooled + n * 384;
    const float* w = W + o * 384;
    for (int k = 0; k < 384; ++k) s += p[k] * w[k];
    out[i] = s;
}

// ---------------------------------------------------------------------------
// Host side
// ---------------------------------------------------------------------------
struct MsgIdx { int A_res, b, bnb, bns, w; };
struct TcnIdx {
    int b1[4], bn1b[4], bn1s[4], bntb[4], bnts[4], bt[4], w1[4], wt[4];
    int mp_b1, mp_bn1b, mp_bn1s, mp_bn2b, mp_bn2s, mp_w1;
    int p1_b, p1_bnb, p1_bns, p1_w;
    int res_b, res_bnb, res_bns, res_w;
    bool has_res;
};

extern "C" void kernel_launch(void* const* d_in, const int* in_sizes, int n_in,
                              void* d_out, int out_size, void* d_ws, size_t ws_size,
                              hipStream_t stream)
{
    (void)in_sizes; (void)n_in; (void)out_size; (void)ws_size;

    // ---- input leaf indices: x first, then params leaves in jax sorted-dict
    // order (lists in order), then A_powers ----
    int idx = 0;
    const int I_x = idx++;
    const int I_db_b = idx++, I_db_s = idx++;
    const int I_fc_b = idx++, I_fc_w = idx++;
    MsgIdx g1, g2, g3;
    auto msg = [&](MsgIdx& g) { g.A_res = idx++; g.b = idx++; g.bnb = idx++; g.bns = idx++; g.w = idx++; };
    msg(g1); msg(g2); msg(g3);
    TcnIdx t1a, t1b, t1c, t2a, t2b, t2c, t3a, t3b, t3c;
    auto tci = [&](TcnIdx& t, bool res) {
        for (int d = 0; d < 4; ++d) {
            t.b1[d] = idx++; t.bn1b[d] = idx++; t.bn1s[d] = idx++;
            t.bntb[d] = idx++; t.bnts[d] = idx++; t.bt[d] = idx++;
            t.w1[d] = idx++; t.wt[d] = idx++;
        }
        t.mp_b1 = idx++; t.mp_bn1b = idx++; t.mp_bn1s = idx++;
        t.mp_bn2b = idx++; t.mp_bn2s = idx++; t.mp_w1 = idx++;
        t.p1_b = idx++; t.p1_bnb = idx++; t.p1_bns = idx++; t.p1_w = idx++;
        t.has_res = res;
        if (res) { t.res_b = idx++; t.res_bnb = idx++; t.res_bns = idx++; t.res_w = idx++; }
        else { t.res_b = t.res_bnb = t.res_bns = t.res_w = 0; }
    };
    tci(t1a, false); tci(t1b, false); tci(t1c, false);
    tci(t2a, true);  tci(t2b, false); tci(t2c, false);
    tci(t3a, true);  tci(t3b, false); tci(t3c, false);
    const int I_Apow = idx++;

    auto F = [&](int i) -> const float* { return (const float*)d_in[i]; };

    // ---- workspace bump allocator (deterministic layout each call) ----
    char* base = (char*)d_ws;
    size_t off = 0;
    auto alloc = [&](size_t bytes) -> void* {
        void* p = base + off;
        off = (off + bytes + 255) & ~(size_t)255;
        return p;
    };
    const long SPW = 720000;  // max per-sample activation floats
    float* actA = (float*)alloc(16L * SPW * 4);
    float* actB = (float*)alloc(16L * SPW * 4);
    u16*   actH = (u16*)  alloc(16L * SPW * 2);     // transposed (S, C) bf16
    float* tmpF = (float*)alloc(16L * 240000 * 4);
    u16*   tmpH = (u16*)  alloc(16L * 260000 * 2);  // transposed (SP, KpT) bf16
    u16*   aggH = (u16*)  alloc(9360000L * 2);      // transposed (S, Kp) bf16
    float* pooled = (float*)alloc(8 * 384 * 4);

    auto packW = [&](const float* w, int Md, int Kd, int kstride, int koff,
                     int& Mp, int& Kp) -> u16* {
        Mp = (Md + 63) & ~63; Kp = (Kd + 31) & ~31;
        u16* p = (u16*)alloc((size_t)Mp * Kp * 2);
        int cnt = Mp * Kp;
        packW_kernel<<<(cnt + 255) / 256, 256, 0, stream>>>(w, p, Md, Kd, Mp, Kp, kstride, koff);
        return p;
    };
    auto gemm = [&](const u16* Wp, const u16* Xh, float* Out,
                    int Md, int Mp, int Kp,
                    int T_out, int ts, int toff_m,
                    long xss, long oss, int nz,
                    const float* bias, const float* bns, const float* bnb, int flags) {
        dim3 g((unsigned)((T_out * 25 + 63) / 64), (unsigned)(Mp / 64), (unsigned)nz);
        gemm_wmma_kernel<<<g, 256, 0, stream>>>(Wp, Xh, Out, Md, Kp,
                                                T_out, ts, toff_m, xss, oss,
                                                bias, bns, bnb, flags);
    };

    // ---- MS-GCN: per-sample aggregation (VALU) + WMMA GEMM over (K*C) ----
    auto msgcn = [&](const MsgIdx& g, const float* inF, float* outF,
                     int cin, int cout, int T) {
        int KC = 13 * cin, Mp, Kp;
        u16* Wp = packW(F(g.w), cout, KC, 1, 0, Mp, Kp);
        long inss = (long)cin * T * 25, outss = (long)cout * T * 25;
        int cnt = Kp * T * 25;
        for (int n = 0; n < 16; ++n) {
            aggT_kernel<<<(cnt + 255) / 256, 256, 0, stream>>>(
                F(I_Apow), F(g.A_res), inF + n * inss, aggH, cin, T, Kp);
            gemm(Wp, aggH, outF + n * outss, cout, Mp, Kp, T, 1, 0, 0, 0, 1,
                 F(g.b), F(g.bns), F(g.bnb), GF_BIAS | GF_BN | GF_RELU);
        }
        long cnt2 = 16L * outss;
        packAT_kernel<<<(unsigned)((cnt2 + 255) / 256), 256, 0, stream>>>(
            outF, actH, cout, T * 25);
    };

    // ---- MS-TCN ----
    auto tcn = [&](const TcnIdx& t, const float* inF, float* outF,
                   int cin, int cout, int T_in, int stride) {
        int T_out = (stride == 2) ? T_in / 2 : T_in;
        int bc = cout / 6;
        int KpT = (bc + 31) & ~31;
        int SP  = (T_in + 8) * 25;       // tap-input padded spatial rows
        long inss = (long)T_in * 25 * cin;       // actH sample stride (S x C)
        long outss = (long)cout * T_out * 25;
        long tss = (long)bc * T_in * 25;
        long tssP = (long)SP * KpT;
        long So = (long)T_out * 25;
        int Mp, Kp, Mpt, Kpt;
        for (int d = 0; d < 4; ++d) {
            u16* W1 = packW(F(t.w1[d]), bc, cin, 1, 0, Mp, Kp);
            gemm(W1, actH, tmpF, bc, Mp, cin, T_in, 1, 0, inss, tss, 16,
                 F(t.b1[d]), F(t.bn1s[d]), F(t.bn1b[d]), GF_BIAS | GF_BN | GF_RELU);
            {
                long cnt = 16L * tssP;
                pack_padT_kernel<<<(unsigned)((cnt + 255) / 256), 256, 0, stream>>>(
                    tmpF, tmpH, bc, KpT, T_in);
            }
            for (int tap = 0; tap < 3; ++tap) {
                u16* Wt = packW(F(t.wt[d]), bc, bc, 3, tap, Mpt, Kpt);
                int fl = (tap == 0) ? GF_BIAS
                       : (tap == 1) ? GF_ACC
                                    : (GF_ACC | GF_ACCFIRST | GF_BN);
                gemm(Wt, tmpH, outF + (long)d * bc * So, bc, Mpt, KpT,
                     T_out, stride, (tap - 1) * (d + 1) + 4, tssP, outss, 16,
                     F(t.bt[d]), F(t.bnts[d]), F(t.bntb[d]), fl);
            }
        }
        {   // maxpool branch
            u16* Wm = packW(F(t.mp_w1), bc, cin, 1, 0, Mp, Kp);
            gemm(Wm, actH, tmpF, bc, Mp, cin, T_in, 1, 0, inss, tss, 16,
                 F(t.mp_b1), F(t.mp_bn1s), F(t.mp_bn1b), GF_BIAS | GF_BN | GF_RELU);
            long cnt = 16L * bc * T_out * 25;
            maxpool_bn_kernel<<<(unsigned)((cnt + 255) / 256), 256, 0, stream>>>(
                tmpF, outF + 4L * bc * So, F(t.mp_bn2s), F(t.mp_bn2b),
                bc, T_in, T_out, stride, tss, outss);
        }
        {   // strided 1x1 branch
            u16* Wq = packW(F(t.p1_w), bc, cin, 1, 0, Mp, Kp);
            gemm(Wq, actH, outF + 5L * bc * So, bc, Mp, cin,
                 T_out, stride, 0, inss, outss, 16,
                 F(t.p1_b), F(t.p1_bns), F(t.p1_bnb), GF_BIAS | GF_BN);
        }
        long cnt = 16L * outss;
        if (t.has_res) {
            u16* Wr = packW(F(t.res_w), cout, cin, 1, 0, Mp, Kp);
            gemm(Wr, actH, outF, cout, Mp, cin, T_out, stride, 0,
                 inss, outss, 16, F(t.res_b), F(t.res_bns), F(t.res_bnb),
                 GF_BIAS | GF_BN | GF_ACC);
            add_relu_kernel<<<(unsigned)((cnt + 255) / 256), 256, 0, stream>>>(
                outF, nullptr, cnt, 1);
        } else {
            add_relu_kernel<<<(unsigned)((cnt + 255) / 256), 256, 0, stream>>>(
                outF, inF, cnt, 1);
        }
        packAT_kernel<<<(unsigned)((cnt + 255) / 256), 256, 0, stream>>>(
            outF, actH, cout, T_out * 25);
    };

    // ---- pipeline ----
    databn_kernel<<<(360000 + 255) / 256, 256, 0, stream>>>(
        F(I_x), F(I_db_s), F(I_db_b), actA);

    // stage 1 (T=300, C 3->96)
    msgcn(g1, actA, actB, 3, 96, 300);
    tcn(t1a, actB, actA, 96, 96, 300, 1);
    tcn(t1b, actA, actB, 96, 96, 300, 1);
    tcn(t1c, actB, actA, 96, 96, 300, 1);
    // stage 2 (T 300->150, C 96->192)
    msgcn(g2, actA, actB, 96, 96, 300);
    tcn(t2a, actB, actA, 96, 192, 300, 2);
    tcn(t2b, actA, actB, 192, 192, 150, 1);
    tcn(t2c, actB, actA, 192, 192, 150, 1);
    // stage 3 (T 150->75, C 192->384)
    msgcn(g3, actA, actB, 192, 192, 150);
    tcn(t3a, actB, actA, 192, 384, 150, 2);
    tcn(t3b, actA, actB, 384, 384, 75, 1);
    tcn(t3c, actB, actA, 384, 384, 75, 1);

    // head: global mean (M,T,V) + FC
    meanpool_kernel<<<8 * 384, 256, 0, stream>>>(actA, pooled, 384, 75 * 25);
    fc_kernel<<<(480 + 63) / 64, 64, 0, stream>>>(
        pooled, F(I_fc_w), F(I_fc_b), (float*)d_out);
}
